// sparseconnect_layer_26637387170397
// MI455X (gfx1250) — compile-verified, matrix operand order checked by reading, not run on hardware
//
#include <hip/hip_runtime.h>

typedef __attribute__((ext_vector_type(16))) _Float16 v16h;
typedef __attribute__((ext_vector_type(8)))  _Float16 v8h;
typedef __attribute__((ext_vector_type(4)))  _Float16 v4h;
typedef __attribute__((ext_vector_type(8)))  float    v8f;
typedef __attribute__((ext_vector_type(4)))  unsigned int v4u;
typedef __attribute__((ext_vector_type(8)))  int      v8i;
typedef __attribute__((ext_vector_type(4)))  int      v4i;

static constexpr int UNITS = 2048;
static constexpr int IN    = 2048;
static constexpr int BATCH = 4096;
static constexpr int KCONN = 16;
static constexpr float GEPS = 1e-20f;

// GEMM tiling
static constexpr int BM = 128;          // block tile M
static constexpr int BN = 128;          // block tile N
static constexpr int BK = 32;           // K step (one 16x16x32 WMMA deep)
static constexpr int LDS_ROW_H = 40;    // 32 halves + 4-dword TDM pad -> 80B stride
static constexpr int TILE_BYTES = BM * LDS_ROW_H * 2;   // 10240 B

union F16Frag { v16h v; v8h h[2]; };

// ---------------------------------------------------------------------------
// Kernel 1: per-unit top-16 of pert = D + gumbel(u). One block per unit.
// ---------------------------------------------------------------------------
__global__ __launch_bounds__(256) void sc_topk_kernel(
    const float* __restrict__ D, const float* __restrict__ U,
    const float* __restrict__ W, int* __restrict__ idxc,
    float* __restrict__ maxes)
{
    __shared__ float pert[IN];
    __shared__ float smax[256];
    __shared__ int   sidx[256];

    const int unit = blockIdx.x;
    const int tid  = threadIdx.x;
    const size_t row = (size_t)unit * IN;

    for (int f = tid; f < IN; f += 256) {
        float un = U[row + f];
        float gn = -0.1f * logf(-logf(un + GEPS) + GEPS);
        pert[f] = D[row + f] + gn;
    }
    __syncthreads();

    float mymax = 0.0f;  // meaningful on tid==0 only
    for (int it = 0; it < KCONN; ++it) {
        float bv = -3.0e38f; int bi = 0;
        for (int f = tid; f < IN; f += 256) {
            float v = pert[f];
            if (v > bv) { bv = v; bi = f; }
        }
        smax[tid] = bv; sidx[tid] = bi;
        __syncthreads();
        for (int s = 128; s > 0; s >>= 1) {
            if (tid < s) {
                if (smax[tid + s] > smax[tid]) {
                    smax[tid] = smax[tid + s];
                    sidx[tid] = sidx[tid + s];
                }
            }
            __syncthreads();
        }
        if (tid == 0) {
            int w = sidx[0];
            idxc[unit * KCONN + it] = w;
            pert[w] = -3.0e38f;
            float aw = fabsf(W[row + w]);
            if (aw > mymax) mymax = aw;
        }
        __syncthreads();
    }
    if (tid == 0) maxes[unit] = mymax;
}

// ---------------------------------------------------------------------------
// Kernel 2: reduce per-unit maxes -> scale = 127.5/max; quantize bias.
// ---------------------------------------------------------------------------
__global__ __launch_bounds__(256) void sc_scale_bias_kernel(
    const float* __restrict__ maxes, const float* __restrict__ b,
    float* __restrict__ scale_out, float* __restrict__ bq)
{
    __shared__ float smax[256];
    const int tid = threadIdx.x;
    float m = 0.0f;
    for (int i = tid; i < UNITS; i += 256) m = fmaxf(m, maxes[i]);
    smax[tid] = m;
    __syncthreads();
    for (int s = 128; s > 0; s >>= 1) {
        if (tid < s) smax[tid] = fmaxf(smax[tid], smax[tid + s]);
        __syncthreads();
    }
    const float scale = 127.5f / fmaxf(smax[0], 1e-30f);
    if (tid == 0) scale_out[0] = scale;
    for (int u = tid; u < UNITS; u += 256)
        bq[u] = __builtin_rintf(scale * b[u]);
}

// ---------------------------------------------------------------------------
// Kernel 3a: build dense quantized f16 weight matrix (16 nonzeros per row).
// ---------------------------------------------------------------------------
__global__ __launch_bounds__(256) void sc_build_wq_kernel(
    const float* __restrict__ W, const int* __restrict__ idxc,
    const float* __restrict__ scale_p, _Float16* __restrict__ wq)
{
    const int unit = blockIdx.x;
    const int tid  = threadIdx.x;
    _Float16* row = wq + (size_t)unit * IN;
    v4h z = {};
    for (int i = tid; i < IN / 4; i += 256) ((v4h*)row)[i] = z;
    __syncthreads();
    if (tid < KCONN) {
        const float scale = scale_p[0];
        const int f = idxc[unit * KCONN + tid];
        row[f] = (_Float16)__builtin_rintf(scale * W[(size_t)unit * IN + f]);
    }
}

// ---------------------------------------------------------------------------
// Kernel 3b: x fp32 -> f16 (vectorized 4-wide).
// ---------------------------------------------------------------------------
__global__ __launch_bounds__(256) void sc_convert_x_kernel(
    const float* __restrict__ x, _Float16* __restrict__ xh)
{
    const int i = (blockIdx.x * 256 + threadIdx.x) * 4;
    const float4 v = *(const float4*)(x + i);
    v4h o = { (_Float16)v.x, (_Float16)v.y, (_Float16)v.z, (_Float16)v.w };
    *(v4h*)(xh + i) = o;
}

// ---------------------------------------------------------------------------
// TDM helper: issue tensor_load_to_lds of a [128 rows x 32 halves] tile from a
// row-major f16 matrix with row stride 2048, padded in LDS to 80B row stride.
// D# packing per CDNA5 ISA. This toolchain's
// builtin takes 6 args: (g0 u32x4, g1 i32x8, g2 i32x4, g3 i32x4, i32x8, cpol).
// ---------------------------------------------------------------------------
__device__ __attribute__((always_inline)) inline void tdm_load_tile(
    const _Float16* gtile, unsigned int lds_off, unsigned int tensor_rows)
{
    const unsigned long long ga = (unsigned long long)(uintptr_t)gtile;
    v4u g0;
    g0.x = 1u;                                   // count=1, user descriptor
    g0.y = lds_off;                              // lds_addr
    g0.z = (unsigned int)ga;                     // global_addr[31:0]
    g0.w = (unsigned int)(ga >> 32) | (2u << 30);// global_addr[56:32] | type=2

    v8i g1;
    // data_size=2B (code 1) | pad_enable | pad_interval=16dw (code 3)
    // | pad_amount=4dw (code 3)
    g1[0] = (1 << 16) | (1 << 20) | (3 << 22) | (3 << 25);
    g1[1] = (int)((unsigned)(IN & 0xFFFF) << 16);        // tensor_dim0 lo16
    g1[2] = (int)((tensor_rows & 0xFFFFu) << 16);        // dim0 hi=0 | tensor_dim1 lo16
    g1[3] = (int)(32u << 16);                            // dim1 hi=0 | tile_dim0=32
    g1[4] = 128;                                         // tile_dim1=128 | tile_dim2=0
    g1[5] = IN;                                          // tensor_dim0_stride lo32
    g1[6] = 0;                                           // stride hi | dim1_stride lo
    g1[7] = 0;                                           // dim1_stride hi

    v4i gz4 = {};
    v8i gz8 = {};
    __builtin_amdgcn_tensor_load_to_lds(g0, g1, gz4, gz4, gz8, 0);
}

// ---------------------------------------------------------------------------
// Kernel 4: y = xh @ wq^T + bq via f16 WMMA, TDM-staged LDS double buffering.
// Block = 256 threads (8 waves), tile 128(M) x 128(N); waves 2(M) x 4(N),
// each wave owns 4x2 16x16 accumulators. Wave 0 drives the TDM; everyone
// syncs on s_wait_tensorcnt + workgroup barriers.
// ---------------------------------------------------------------------------
__global__ __launch_bounds__(256) void sc_gemm_wmma_kernel(
    const _Float16* __restrict__ xh, const _Float16* __restrict__ wq,
    const float* __restrict__ bq, float* __restrict__ y)
{
    __shared__ __align__(16) unsigned char smA[2][TILE_BYTES];
    __shared__ __align__(16) unsigned char smB[2][TILE_BYTES];

    const int lane  = threadIdx.x & 31;
    const int wave  = threadIdx.x >> 5;
    const int waveM = wave & 1;        // 0..1
    const int waveN = wave >> 1;       // 0..3
    const int lmod  = lane & 15;
    const int lhalf = lane >> 4;
    const int kbyte = lhalf * 16;      // 8 halves sub-offset in bytes

    const int mblk = blockIdx.x * BM;
    const int nblk = blockIdx.y * BN;

    const _Float16* gA = xh + (size_t)mblk * IN;   // tile row 0, k advances
    const _Float16* gB = wq + (size_t)nblk * IN;

    v8f zero = {};
    v8f acc[4][2];
#pragma unroll
    for (int i = 0; i < 4; ++i)
#pragma unroll
        for (int j = 0; j < 2; ++j) acc[i][j] = zero;

    // Per-lane LDS fragment base addresses (padded 80B row stride).
    const unsigned char* aBase[4];
    const unsigned char* bBase[2];
#pragma unroll
    for (int i = 0; i < 4; ++i)
        aBase[i] = &smA[0][0] + (waveM * 64 + i * 16 + lmod) * (LDS_ROW_H * 2) + kbyte;
#pragma unroll
    for (int j = 0; j < 2; ++j)
        bBase[j] = &smB[0][0] + (waveN * 32 + j * 16 + lmod) * (LDS_ROW_H * 2) + kbyte;
    const int bufStrideA = (int)(&smA[1][0] - &smA[0][0]);
    const int bufStrideB = (int)(&smB[1][0] - &smB[0][0]);

    const int NK = IN / BK;            // 64 K-steps

    if (wave == 0) {
        tdm_load_tile(gA, (unsigned int)(uintptr_t)&smA[0][0], BATCH);
        tdm_load_tile(gB, (unsigned int)(uintptr_t)&smB[0][0], UNITS);
    }

    for (int it = 0; it < NK; ++it) {
        const int cur = it & 1;
        if (wave == 0) {
            if (it + 1 < NK) {
                const int nxt = cur ^ 1;
                tdm_load_tile(gA + (it + 1) * BK,
                              (unsigned int)(uintptr_t)&smA[nxt][0], BATCH);
                tdm_load_tile(gB + (it + 1) * BK,
                              (unsigned int)(uintptr_t)&smB[nxt][0], UNITS);
                __builtin_amdgcn_s_wait_tensorcnt(2);  // current buffer complete
            } else {
                __builtin_amdgcn_s_wait_tensorcnt(0);
            }
        }
        __syncthreads();   // current LDS tiles ready for all waves

        F16Frag a[4], b[2];
#pragma unroll
        for (int i = 0; i < 4; ++i) {
            const unsigned char* p = aBase[i] + cur * bufStrideA;
            a[i].h[0] = *(const v8h*)(p);
            a[i].h[1] = *(const v8h*)(p + 32);
        }
#pragma unroll
        for (int j = 0; j < 2; ++j) {
            const unsigned char* p = bBase[j] + cur * bufStrideB;
            b[j].h[0] = *(const v8h*)(p);
            b[j].h[1] = *(const v8h*)(p + 32);
        }
#pragma unroll
        for (int i = 0; i < 4; ++i)
#pragma unroll
            for (int j = 0; j < 2; ++j)
                acc[i][j] = __builtin_amdgcn_wmma_f32_16x16x32_f16(
                    false, a[i].v, false, b[j].v,
                    (short)0, acc[i][j], false, false);

        __syncthreads();   // all waves done reading before TDM overwrites
    }

    // C/D layout: lanes 0-15 -> rows v+0, col=lane; lanes 16-31 -> rows v+8.
    const int m0 = mblk + waveM * 64;
    const int n0 = nblk + waveN * 32;
#pragma unroll
    for (int i = 0; i < 4; ++i) {
#pragma unroll
        for (int j = 0; j < 2; ++j) {
            const int col  = n0 + j * 16 + lmod;
            const float bb = bq[col];
#pragma unroll
            for (int v = 0; v < 8; ++v) {
                const int row = m0 + i * 16 + v + lhalf * 8;
                y[(size_t)row * UNITS + col] = acc[i][j][v] + bb;
            }
        }
    }
}

// ---------------------------------------------------------------------------
// Launch
// ---------------------------------------------------------------------------
extern "C" void kernel_launch(void* const* d_in, const int* in_sizes, int n_in,
                              void* d_out, int out_size, void* d_ws, size_t ws_size,
                              hipStream_t stream) {
    const float* x = (const float*)d_in[0];   // [4096, 2048]
    const float* W = (const float*)d_in[1];   // [2048, 2048]
    const float* b = (const float*)d_in[2];   // [2048]
    const float* D = (const float*)d_in[3];   // [2048, 2048]
    const float* U = (const float*)d_in[4];   // [1, 2048, 2048]
    float* y = (float*)d_out;                 // [4096, 2048]

    char* ws = (char*)d_ws;
    int*      idxc  = (int*)(ws + 0);                       // 128 KB
    float*    maxes = (float*)(ws + (128u << 10));          // 8 KB
    float*    scale = (float*)(ws + (144u << 10));          // 4 B
    float*    bq    = (float*)(ws + (152u << 10));          // 8 KB
    _Float16* xh    = (_Float16*)(ws + (1u << 20));         // 16 MB
    _Float16* wq    = (_Float16*)(ws + (20u << 20));        // 8 MB

    sc_topk_kernel<<<UNITS, 256, 0, stream>>>(D, U, W, idxc, maxes);
    sc_scale_bias_kernel<<<1, 256, 0, stream>>>(maxes, b, scale, bq);
    sc_build_wq_kernel<<<UNITS, 256, 0, stream>>>(W, idxc, scale, wq);
    sc_convert_x_kernel<<<(BATCH * IN) / (256 * 4), 256, 0, stream>>>(x, xh);
    sc_gemm_wmma_kernel<<<dim3(BATCH / BM, UNITS / BN), 256, 0, stream>>>(xh, wq, bq, y);
}
